// CRFLayer_82703890252445
// MI455X (gfx1250) — matrix operand order repaired and checked by reading, not compiled, not run on hardware
//
#include <hip/hip_runtime.h>
#include <hip/hip_bf16.h>
#include <math.h>

// CRF log-likelihood on MI455X (gfx1250, wave32).
//
// Forward-scan recurrence recast as a per-timestep GEMM:
//   alpha_t = logit_t + m + log( exp(alpha_{t-1} - m) @ exp(trans) )
// exp(trans) (48x48, padded to 64x48) is precomputed once in bf16 and kept
// resident in VGPRs as six v_wmma_f32_16x16x32_bf16 B-operands for the whole
// 2048-step scan. Each 16-batch tile is one wave32 workgroup (16 tiles).
// Per step: 6 WMMAs + 32 exp + cross-lane max. logits (101 MB) streamed with
// global_prefetch_b8 issued 8 steps ahead.

typedef __attribute__((ext_vector_type(16))) __bf16 v16bf;
typedef __attribute__((ext_vector_type(8)))  float  v8f;

#define B_TOT 256
#define T_LEN 2048
#define K_DIM 48

// ---------------------------------------------------------------------------
// Kernel A: unary + binary path score per batch (gather + reduce).
// ---------------------------------------------------------------------------
__global__ __launch_bounds__(256)
void crf_scores_kernel(const float* __restrict__ logits,
                       const float* __restrict__ trans,
                       const int*   __restrict__ targets,
                       const int*   __restrict__ lengths,
                       float*       __restrict__ scores) {
    const int b   = blockIdx.x;
    const int len = lengths[b];
    const int*   tg = targets + (size_t)b * T_LEN;
    const float* lg = logits  + (size_t)b * T_LEN * K_DIM;

    float s = 0.f;
    for (int t = threadIdx.x; t < len; t += blockDim.x) {
        const int k = tg[t];
        s += lg[(size_t)t * K_DIM + k];             // unary, masked t < len
        if (t >= 1) s += trans[tg[t - 1] * K_DIM + k]; // binary, masked t < len
    }
    __shared__ float red[8];
    #pragma unroll
    for (int off = 16; off > 0; off >>= 1) s += __shfl_xor(s, off, 32);
    if ((threadIdx.x & 31) == 0) red[threadIdx.x >> 5] = s;
    __syncthreads();
    if (threadIdx.x == 0) {
        float tot = 0.f;
        #pragma unroll
        for (int w = 0; w < 8; ++w) tot += red[w];
        scores[b] = tot;
    }
}

// ---------------------------------------------------------------------------
// Kernel B: WMMA forward scan. One wave32 per 16-batch tile.
// ---------------------------------------------------------------------------
__global__ __launch_bounds__(32)
void crf_scan_kernel(const float* __restrict__ logits,
                     const float* __restrict__ trans,
                     const int*   __restrict__ lengths,
                     const float* __restrict__ scores,
                     float*       __restrict__ out) {
    const int tile = blockIdx.x;
    const int lane = threadIdx.x;      // 0..31
    const int half = lane >> 4;        // lane-half (ISA A/B layout split)
    const int l16  = lane & 15;
    const int b0   = tile * 16;

    __shared__ float alphaLds[16][K_DIM];
    __shared__ float mLds[16];
    __shared__ float lnLds[16];
    __shared__ int   lastLds[16];
    __shared__ int   lenLds[16];

    // --- Precompute E = exp(trans) as bf16 WMMA B-operands (32x16 tiles). ---
    // B layout (16-bit, 32x16): lane holds column j = lane%16; element e holds
// row i = 32*chunk + (e&7) + 16*(e/8) + 8*(lane/16). Rows >= 48 are zero padding.
    auto packE = [&](int n, int c) {
        v16bf e;
        const int j = n * 16 + l16;
        #pragma unroll
        for (int q = 0; q < 16; ++q) {
            const int i = c * 32 + (q & 7) + 16 * (q >> 3) + 8 * half;
            const float v = (i < K_DIM) ? __expf(trans[(size_t)i * K_DIM + j]) : 0.f;
            e[q] = (__bf16)v;
        }
        return e;
    };
    const v16bf E00 = packE(0, 0), E01 = packE(0, 1);
    const v16bf E10 = packE(1, 0), E11 = packE(1, 1);
    const v16bf E20 = packE(2, 0), E21 = packE(2, 1);

    // --- Per-batch metadata + alpha0 = logits[:,0,:]. ---
    if (lane < 16) {
        const int len = lengths[b0 + lane];
        lenLds[lane]  = len;
        lastLds[lane] = (len - 1 > 0) ? (len - 1) : 0;
        lnLds[lane]   = 0.f;
    }
    for (int idx = lane; idx < 16 * K_DIM; idx += 32) {
        const int b = idx / K_DIM, k = idx % K_DIM;
        alphaLds[b][k] = logits[((size_t)(b0 + b) * T_LEN) * K_DIM + k];
    }
    __syncthreads();

    if (lane < 16) {
        float m = -3.402823466e38f;
        #pragma unroll 4
        for (int k = 0; k < K_DIM; ++k) m = fmaxf(m, alphaLds[lane][k]);
        mLds[lane] = m;
        if (lenLds[lane] > 0 && lastLds[lane] == 0) {   // length-1 sequences
            float s = 0.f;
            for (int k = 0; k < K_DIM; ++k) s += __expf(alphaLds[lane][k] - m);
            lnLds[lane] = m + __logf(s);
        }
    }
    __syncthreads();

    // Uniform loop bound for this tile: max last-index over live batches.
    int tm = (lane < 16 && lenLds[lane] > 0) ? lastLds[lane] : 0;
    #pragma unroll
    for (int off = 16; off > 0; off >>= 1) {
        const int o = __shfl_xor(tm, off, 32);
        tm = (o > tm) ? o : tm;
    }
    const int tMax = tm;

    for (int t = 1; t <= tMax; ++t) {
        // Prefetch logits 8 steps ahead (emits global_prefetch_b8).
        const int tp = t + 8;
        if (tp <= tMax) {
            __builtin_prefetch(
                logits + ((size_t)(b0 + l16) * T_LEN + tp) * K_DIM + half * 32, 0, 3);
        }

        // --- Pack A = bf16(exp(alpha - m)) in the ISA 16-bit A layout. ---
        // lane holds batch row l16; element e -> K index:
        //   k = 16*(e/8) + 2*((e/2)&3) + (e&1) + 8*half   (operand 1: +32)
        const float mb = mLds[l16];
        v16bf A0, A1;
        #pragma unroll
        for (int e = 0; e < 16; ++e) {
            const int k = ((e >> 3) << 4) + (((e >> 1) & 3) << 1) + (e & 1) + half * 8;
            A0[e] = (__bf16)__expf(alphaLds[l16][k] - mb);
            const int k1 = k + 32;
            A1[e] = (k1 < K_DIM) ? (__bf16)__expf(alphaLds[l16][k1] - mb) : (__bf16)0.f;
        }

        // --- Load this step's logits in the f32 C/D layout. ---
        // lane holds column j = l16 (+16 per N-tile), VGPR slot v -> batch v+8*half.
        float lg0[8], lg1[8], lg2[8], mrow[8];
        #pragma unroll
        for (int v = 0; v < 8; ++v) {
            const int b = v + half * 8;
            const float* row = logits + ((size_t)(b0 + b) * T_LEN + t) * K_DIM;
            lg0[v] = row[l16];
            lg1[v] = row[16 + l16];
            lg2[v] = row[32 + l16];
            mrow[v] = mLds[b];
        }

        // --- 6 WMMAs: acc_n = U(16x64) @ E(64x16-tile n), f32 accumulate. ---
        v8f acc0 = {}, acc1 = {}, acc2 = {};
        acc0 = __builtin_amdgcn_wmma_f32_16x16x32_bf16(false, A0, false, E00, (short)0, acc0, false, false);
        acc0 = __builtin_amdgcn_wmma_f32_16x16x32_bf16(false, A1, false, E01, (short)0, acc0, false, false);
        acc1 = __builtin_amdgcn_wmma_f32_16x16x32_bf16(false, A0, false, E10, (short)0, acc1, false, false);
        acc1 = __builtin_amdgcn_wmma_f32_16x16x32_bf16(false, A1, false, E11, (short)0, acc1, false, false);
        acc2 = __builtin_amdgcn_wmma_f32_16x16x32_bf16(false, A0, false, E20, (short)0, acc2, false, false);
        acc2 = __builtin_amdgcn_wmma_f32_16x16x32_bf16(false, A1, false, E21, (short)0, acc2, false, false);

        // --- new_alpha = logit + m + log(acc); track per-batch row max. ---
        float na0[8], na1[8], na2[8], nmax[8];
        #pragma unroll
        for (int v = 0; v < 8; ++v) {
            const float a0 = lg0[v] + mrow[v] + __logf(acc0[v]);
            const float a1 = lg1[v] + mrow[v] + __logf(acc1[v]);
            const float a2 = lg2[v] + mrow[v] + __logf(acc2[v]);
            na0[v] = a0; na1[v] = a1; na2[v] = a2;
            nmax[v] = fmaxf(fmaxf(a0, a1), a2);
        }
        // Row max across the 16 columns held by this lane-half.
        #pragma unroll
        for (int off = 1; off < 16; off <<= 1) {
            #pragma unroll
            for (int v = 0; v < 8; ++v)
                nmax[v] = fmaxf(nmax[v], __shfl_xor(nmax[v], off, 16));
        }

        __syncthreads();                       // all reads of old alpha done
        #pragma unroll
        for (int v = 0; v < 8; ++v) {
            const int b = v + half * 8;
            alphaLds[b][l16]      = na0[v];
            alphaLds[b][16 + l16] = na1[v];
            alphaLds[b][32 + l16] = na2[v];
        }
        if (l16 == 0) {
            #pragma unroll
            for (int v = 0; v < 8; ++v) mLds[v + half * 8] = nmax[v];
        }
        __syncthreads();

        // Capture log_norm for batches finishing at this step.
        if (lane < 16 && lenLds[lane] > 0 && lastLds[lane] == t) {
            const float m = mLds[lane];
            float s = 0.f;
            for (int k = 0; k < K_DIM; ++k) s += __expf(alphaLds[lane][k] - m);
            lnLds[lane] = m + __logf(s);
        }
        __syncthreads();
    }

    if (lane < 16) {
        const int b = b0 + lane;
        const float ln = (lenLds[lane] > 0) ? lnLds[lane] : 0.f;
        out[b] = scores[b] - ln;
    }
}

// ---------------------------------------------------------------------------
extern "C" void kernel_launch(void* const* d_in, const int* in_sizes, int n_in,
                              void* d_out, int out_size, void* d_ws, size_t ws_size,
                              hipStream_t stream) {
    const float* logits  = (const float*)d_in[0];
    const float* trans   = (const float*)d_in[1];
    const int*   targets = (const int*)d_in[2];
    const int*   lengths = (const int*)d_in[3];
    float*       out     = (float*)d_out;
    float*       scores  = (float*)d_ws;   // 256 floats of scratch

    crf_scores_kernel<<<B_TOT, 256, 0, stream>>>(logits, trans, targets, lengths, scores);
    crf_scan_kernel<<<B_TOT / 16, 32, 0, stream>>>(logits, trans, lengths, scores, out);
}
